// MBI_16509854286301
// MI455X (gfx1250) — matrix operand order repaired and checked by reading, not compiled, not run on hardware
//
#include <hip/hip_runtime.h>
#include <cstdint>

// Problem constants (match reference)
#define N_NODES 20000
#define N_EDGES 320000
#define HID     128
#define NHEAD   8
#define HDIM    16

typedef __attribute__((ext_vector_type(16))) __bf16 v16bf;
typedef __attribute__((ext_vector_type(8)))  __bf16 v8bf;
typedef __attribute__((ext_vector_type(4)))  __bf16 v4bf;
typedef __attribute__((ext_vector_type(2)))  __bf16 v2bf;
typedef __attribute__((ext_vector_type(8)))  float  v8f;
typedef __attribute__((ext_vector_type(4)))  unsigned int v4u;
typedef __attribute__((ext_vector_type(8)))  int    v8i;
typedef __attribute__((ext_vector_type(4)))  int    v4i;

__device__ __forceinline__ float silu_f(float x) { return x / (1.0f + __expf(-x)); }

// ---------------------------------------------------------------------------
// LayerNorm: one wave per row of 128; output stored directly as bf16 (it is
// consumed only as a WMMA A-operand, which is bf16 anyway).
// ---------------------------------------------------------------------------
__global__ __launch_bounds__(256) void k_layernorm(
    const float* __restrict__ x, const float* __restrict__ gamma,
    const float* __restrict__ beta, __bf16* __restrict__ y) {
  int row  = blockIdx.x * 8 + (threadIdx.x >> 5);
  int lane = threadIdx.x & 31;
  if (row >= N_NODES) return;
  float4 v = reinterpret_cast<const float4*>(x + (size_t)row * HID)[lane];
  float s  = v.x + v.y + v.z + v.w;
  float ss = v.x * v.x + v.y * v.y + v.z * v.z + v.w * v.w;
  for (int m = 16; m >= 1; m >>= 1) {
    s  += __shfl_xor(s,  m, 32);
    ss += __shfl_xor(ss, m, 32);
  }
  float mu  = s * (1.0f / HID);
  float var = ss * (1.0f / HID) - mu * mu;
  float inv = rsqrtf(var + 1e-5f);
  int c = lane * 4;
  v4bf o = {(__bf16)((v.x - mu) * inv * gamma[c + 0] + beta[c + 0]),
            (__bf16)((v.y - mu) * inv * gamma[c + 1] + beta[c + 1]),
            (__bf16)((v.z - mu) * inv * gamma[c + 2] + beta[c + 2]),
            (__bf16)((v.w - mu) * inv * gamma[c + 3] + beta[c + 3])};
  *reinterpret_cast<v4bf*>(y + (size_t)row * HID + c) = o;
}

// ---------------------------------------------------------------------------
// TDM helper: DMA a 128-row x 128-col bf16 tile (row stride 128 elements in
// global) into LDS with 16-B padding every 256-B row (LDS stride 272 B).
// tensor_dim1 = remaining rows -> TDM zero-fills out-of-bounds rows.
// NOTE: compile-only environment; lds_byte_off derived by pointer cast.
// ---------------------------------------------------------------------------
#if defined(__HIP_DEVICE_COMPILE__)
__device__ __forceinline__ void tdm_load_tile_bf16(
    const __bf16* gsrc, unsigned lds_byte_off, unsigned rows_rem) {
  unsigned long long ga = (unsigned long long)(size_t)gsrc;
  v4u g0;
  g0[0] = 1u;                                          // count=1, user mode
  g0[1] = lds_byte_off;                                // LDS dest (bytes)
  g0[2] = (unsigned)(ga & 0xffffffffu);                // global addr lo
  g0[3] = (unsigned)((ga >> 32) & 0x1ffffffu) | (2u << 30);  // hi | type=2
  v8i g1;
  g1[0] = (int)((1u << 16)      // data_size = 2 bytes
              | (1u << 20)      // pad_enable
              | (5u << 22)      // pad_interval: 64 dwords (one 256-B row)
              | (3u << 25));    // pad_amount: 4 dwords (16 B)
  g1[1] = (int)(128u << 16);                           // tensor_dim0 = 128
  g1[2] = (int)((rows_rem & 0xffffu) << 16);           // tensor_dim1 lo16
  g1[3] = (int)((rows_rem >> 16) | (128u << 16));      // dim1 hi16|tile_dim0
  g1[4] = (int)128u;                                   // tile_dim1 = 128 rows
  g1[5] = (int)128u;                                   // tensor_dim0_stride
  g1[6] = 0;
  g1[7] = 0;
  v4i gz = {0, 0, 0, 0};
#if __clang_major__ >= 23
  v8i gz8 = {0, 0, 0, 0, 0, 0, 0, 0};
  __builtin_amdgcn_tensor_load_to_lds(g0, g1, gz, gz, gz8, 0);
#else
  __builtin_amdgcn_tensor_load_to_lds(g0, g1, gz, gz, 0);
#endif
  __builtin_amdgcn_s_wait_tensorcnt(0);
}
#endif

// ---------------------------------------------------------------------------
// Shared GEMM core: computes a 128 x HOUT tile from sA (bf16, stride 136)
// against W chunks staged transposed in sWt.  Fragments via ds_load_b128.
// ---------------------------------------------------------------------------
union Frag16 { v16bf v; v8bf h[2]; };

template <int HOUT, bool BIAS, bool SILU>
__device__ __forceinline__ void gemm_core(
    __bf16 (&sA)[128][HID + 8], __bf16 (&sWt)[64][HID + 8],
    const float* __restrict__ W, const float* __restrict__ bias,
    float* __restrict__ C, int M, int m0) {
  const int tid  = threadIdx.x;
  const int wave = tid >> 5, lane = tid & 31;
  const int lrow = lane & 15, khalf = lane >> 4;
  const int r0   = wave * 16;

  // A fragments: per 16-bit A layout (ISA 7.12.2) a lane's 16 values are two
  // contiguous 8-element K-runs -> two 16-B LDS loads each.
  Frag16 af[4];
#pragma unroll
  for (int t = 0; t < 4; ++t) {
    int k1 = t * 32 + khalf * 8;
    af[t].h[0] = *reinterpret_cast<const v8bf*>(&sA[r0 + lrow][k1]);
    af[t].h[1] = *reinterpret_cast<const v8bf*>(&sA[r0 + lrow][k1 + 16]);
  }

  const v8f vzero = {0.f, 0.f, 0.f, 0.f, 0.f, 0.f, 0.f, 0.f};

  for (int nc = 0; nc < HOUT; nc += 64) {
    __syncthreads();
    // Stage W chunk transposed: sWt[col][k].  Consecutive tids take
    // consecutive columns -> coalesced global reads; packed 2-wide stores.
    for (int it = tid; it < 64 * (HID / 2); it += 256) {
      int col = it & 63;
      int kp  = it >> 6;
      float g0 = W[(size_t)(2 * kp + 0) * HOUT + nc + col];
      float g1 = W[(size_t)(2 * kp + 1) * HOUT + nc + col];
      v2bf p = {(__bf16)g0, (__bf16)g1};
      *reinterpret_cast<v2bf*>(&sWt[col][2 * kp]) = p;
    }
    __syncthreads();

    v8f acc[4];
#pragma unroll
    for (int nt = 0; nt < 4; ++nt) acc[nt] = vzero;

#pragma unroll
    for (int t = 0; t < 4; ++t) {
#pragma unroll
      for (int nt = 0; nt < 4; ++nt) {
        int col = nt * 16 + lrow;
        int kb  = t * 32 + khalf * 8;
        Frag16 bf;
        bf.h[0] = *reinterpret_cast<const v8bf*>(&sWt[col][kb]);
        bf.h[1] = *reinterpret_cast<const v8bf*>(&sWt[col][kb + 16]);
        acc[nt] = __builtin_amdgcn_wmma_f32_16x16x32_bf16(
            false, af[t].v, false, bf.v, (short)0, acc[nt], false, false);
      }
    }

    // Epilogue: bias + optional SiLU, write f32
#pragma unroll
    for (int nt = 0; nt < 4; ++nt) {
      int ocol = nc + nt * 16 + lrow;
      float b = BIAS ? bias[ocol] : 0.0f;
#pragma unroll
      for (int r = 0; r < 8; ++r) {
        int orow = m0 + r0 + khalf * 8 + r;
        if (orow < M) {
          float v = acc[nt][r] + b;
          if (SILU) v = silu_f(v);
          C[(size_t)orow * HOUT + ocol] = v;
        }
      }
    }
  }
}

// GEMM with fp32 A (inputs/accumulators): convert while staging.
template <int HOUT, bool BIAS, bool SILU>
__global__ __launch_bounds__(256) void k_gemm(
    const float* __restrict__ A, const float* __restrict__ W,
    const float* __restrict__ bias, float* __restrict__ C, int M) {
  __shared__ __bf16 sA[128][HID + 8];   // 34816 B
  __shared__ __bf16 sWt[64][HID + 8];   // 17408 B
  const int tid = threadIdx.x;
  const int m0  = blockIdx.x * 128;
  for (int it = tid; it < 128 * (HID / 4); it += 256) {
    int r = it / (HID / 4);
    int c = (it % (HID / 4)) * 4;
    float4 v = make_float4(0.f, 0.f, 0.f, 0.f);
    if (m0 + r < M)
      v = *reinterpret_cast<const float4*>(A + (size_t)(m0 + r) * HID + c);
    v4bf p = {(__bf16)v.x, (__bf16)v.y, (__bf16)v.z, (__bf16)v.w};
    *reinterpret_cast<v4bf*>(&sA[r][c]) = p;
  }
  __syncthreads();
  gemm_core<HOUT, BIAS, SILU>(sA, sWt, W, bias, C, M, m0);
}

// GEMM with bf16 A (self-produced intermediates): A tile DMA'd by the
// Tensor Data Mover straight into padded LDS; TENSORcnt-synchronized.
template <int HOUT, bool BIAS, bool SILU>
__global__ __launch_bounds__(256) void k_gemm_bA(
    const __bf16* __restrict__ A, const float* __restrict__ W,
    const float* __restrict__ bias, float* __restrict__ C, int M) {
  __shared__ __bf16 sA[128][HID + 8];
  __shared__ __bf16 sWt[64][HID + 8];
  const int tid = threadIdx.x;
  const int m0  = blockIdx.x * 128;
#if defined(__HIP_DEVICE_COMPILE__)
  if (tid == 0) {
    tdm_load_tile_bf16(A + (size_t)m0 * HID,
                       (unsigned)(size_t)(void*)&sA[0][0],
                       (unsigned)(M - m0));
  }
#else
  for (int it = tid; it < 128 * (HID / 4); it += 256) {
    int r = it / (HID / 4);
    int c = (it % (HID / 4)) * 4;
    if (m0 + r < M)
      *reinterpret_cast<v4bf*>(&sA[r][c]) =
          *reinterpret_cast<const v4bf*>(A + (size_t)(m0 + r) * HID + c);
  }
#endif
  __syncthreads();
  gemm_core<HOUT, BIAS, SILU>(sA, sWt, W, bias, C, M, m0);
}

// ---------------------------------------------------------------------------
// Seed accumulators with residual inputs (re-run every launch: deterministic)
// ---------------------------------------------------------------------------
__global__ void k_init(const float* __restrict__ s_in, const float* __restrict__ v_in,
                       float* __restrict__ s_new, float* __restrict__ v_new) {
  size_t stride = (size_t)gridDim.x * blockDim.x;
  size_t base   = (size_t)blockIdx.x * blockDim.x + threadIdx.x;
  const size_t n1 = (size_t)N_NODES * HID;
  const size_t n2 = (size_t)N_NODES * 3 * HID;
  for (size_t i = base; i < n1; i += stride) s_new[i] = s_in[i];
  for (size_t i = base; i < n2; i += stride) v_new[i] = v_in[i];
}

// ---------------------------------------------------------------------------
// Attention + scalar message, wave per edge; lane owns 4 contiguous channels
// so each head (16 ch) spans a 4-lane group -> 2-step shfl_xor reduction.
// msg stored as bf16 (consumed only as WMMA A-operand); atomics stay fp32.
// ---------------------------------------------------------------------------
__global__ __launch_bounds__(256) void k_attn_msg(
    const int* __restrict__ ei, const float* __restrict__ dist,
    const float* __restrict__ ns, const float* __restrict__ ef,
    const float* __restrict__ alpha, __bf16* __restrict__ msgbf,
    float* __restrict__ s_acc) {
  int e = blockIdx.x * 8 + (threadIdx.x >> 5);
  if (e >= N_EDGES) return;
  int lane = threadIdx.x & 31;
  int i = ei[e], j = ei[N_EDGES + e];
  int c0 = lane * 4;
  float4 nsi = *reinterpret_cast<const float4*>(ns + (size_t)i * HID + c0);
  float4 nsj = *reinterpret_cast<const float4*>(ns + (size_t)j * HID + c0);
  float4 efv = *reinterpret_cast<const float4*>(ef + (size_t)e * HID + c0);
  float4 al  = *reinterpret_cast<const float4*>(alpha + c0);
  float partial = silu_f(nsi.x + nsj.x + efv.x) * al.x
                + silu_f(nsi.y + nsj.y + efv.y) * al.y
                + silu_f(nsi.z + nsj.z + efv.z) * al.z
                + silu_f(nsi.w + nsj.w + efv.w) * al.w;
  partial += __shfl_xor(partial, 1, 32);   // reduce within 4-lane head group
  partial += __shfl_xor(partial, 2, 32);
  float d   = dist[e];
  float cut = 0.5f * (__cosf(3.14159265358979323846f * d / 5.0f) + 1.0f) *
              (d < 5.0f ? 1.0f : 0.0f);
  float attn = partial * cut;
  float mx = nsj.x * efv.x * attn;
  float my = nsj.y * efv.y * attn;
  float mz = nsj.z * efv.z * attn;
  float mw = nsj.w * efv.w * attn;
  v4bf mb = {(__bf16)mx, (__bf16)my, (__bf16)mz, (__bf16)mw};
  *reinterpret_cast<v4bf*>(msgbf + (size_t)e * HID + c0) = mb;
  float* dst = s_acc + (size_t)i * HID + c0;
  atomicAdd(dst + 0, mx);
  atomicAdd(dst + 1, my);
  atomicAdd(dst + 2, mz);
  atomicAdd(dst + 3, mw);
}

// ---------------------------------------------------------------------------
// Edge update: sum_phi = dot(cross(nv_i, ev), cross(nv_j, ev)) per channel.
// ---------------------------------------------------------------------------
__global__ __launch_bounds__(256) void k_edge_update(
    const int* __restrict__ ei, const float* __restrict__ evec,
    const float* __restrict__ nv, const float* __restrict__ fgate,
    const float* __restrict__ edge_feats, float* __restrict__ eout) {
  size_t idx = (size_t)blockIdx.x * 256 + threadIdx.x;
  if (idx >= (size_t)N_EDGES * HID) return;
  int e = (int)(idx / HID), c = (int)(idx % HID);
  int i = ei[e], j = ei[N_EDGES + e];
  float e0 = evec[(size_t)e * 3 + 0];
  float e1 = evec[(size_t)e * 3 + 1];
  float e2 = evec[(size_t)e * 3 + 2];
  float ai0 = nv[((size_t)i * 3 + 0) * HID + c];
  float ai1 = nv[((size_t)i * 3 + 1) * HID + c];
  float ai2 = nv[((size_t)i * 3 + 2) * HID + c];
  float aj0 = nv[((size_t)j * 3 + 0) * HID + c];
  float aj1 = nv[((size_t)j * 3 + 1) * HID + c];
  float aj2 = nv[((size_t)j * 3 + 2) * HID + c];
  float ci0 = ai1 * e2 - ai2 * e1, ci1 = ai2 * e0 - ai0 * e2, ci2 = ai0 * e1 - ai1 * e0;
  float cj0 = aj1 * e2 - aj2 * e1, cj1 = aj2 * e0 - aj0 * e2, cj2 = aj0 * e1 - aj1 * e0;
  float sp = ci0 * cj0 + ci1 * cj1 + ci2 * cj2;
  eout[idx] = edge_feats[idx] + fgate[idx] * sp;
}

// ---------------------------------------------------------------------------
// Vector message scatter: vmsg = nvec[j]*s1 + s2*edge_vector, atomics into i.
// ---------------------------------------------------------------------------
__global__ __launch_bounds__(256) void k_vmsg(
    const int* __restrict__ ei, const float* __restrict__ nvec,
    const float* __restrict__ evec, const float* __restrict__ pm,
    float* __restrict__ v_acc) {
  int e = blockIdx.x * 8 + (threadIdx.x >> 5);
  if (e >= N_EDGES) return;
  int lane = threadIdx.x & 31;
  int c0 = lane * 4;
  int i = ei[e], j = ei[N_EDGES + e];
  float4 s1 = *reinterpret_cast<const float4*>(pm + (size_t)e * 256 + c0);
  float4 s2 = *reinterpret_cast<const float4*>(pm + (size_t)e * 256 + 128 + c0);
  float ev[3] = {evec[(size_t)e * 3 + 0], evec[(size_t)e * 3 + 1], evec[(size_t)e * 3 + 2]};
#pragma unroll
  for (int d = 0; d < 3; ++d) {
    float4 nj = *reinterpret_cast<const float4*>(nvec + ((size_t)j * 3 + d) * HID + c0);
    float* dst = v_acc + ((size_t)i * 3 + d) * HID + c0;
    atomicAdd(dst + 0, nj.x * s1.x + s2.x * ev[d]);
    atomicAdd(dst + 1, nj.y * s1.y + s2.y * ev[d]);
    atomicAdd(dst + 2, nj.z * s1.z + s2.z * ev[d]);
    atomicAdd(dst + 3, nj.w * s1.w + s2.w * ev[d]);
  }
}

// ---------------------------------------------------------------------------
// Final node update from vv (v1|v2) and pp (p1|p2|p3).
// ---------------------------------------------------------------------------
__global__ __launch_bounds__(256) void k_node_final(
    const float* __restrict__ s_new, const float* __restrict__ v_new,
    const float* __restrict__ vv, const float* __restrict__ pp,
    float* __restrict__ out_s, float* __restrict__ out_v) {
  size_t idx = (size_t)blockIdx.x * 256 + threadIdx.x;
  if (idx >= (size_t)N_NODES * HID) return;
  int n = (int)(idx / HID), c = (int)(idx % HID);
  float v1[3], v2[3];
#pragma unroll
  for (int d = 0; d < 3; ++d) {
    v1[d] = vv[((size_t)n * 3 + d) * 256 + c];
    v2[d] = vv[((size_t)n * 3 + d) * 256 + 128 + c];
  }
  float tri = v1[0] * v2[0] + v1[1] * v2[1] + v1[2] * v2[2];
  float nrm = sqrtf(v2[0] * v2[0] + v2[1] * v2[1] + v2[2] * v2[2] + 1e-8f);
  float qua = nrm * nrm * nrm;
  float p1 = pp[(size_t)n * 384 + c];
  float p2 = pp[(size_t)n * 384 + 128 + c];
  float p3 = pp[(size_t)n * 384 + 256 + c];
  out_s[idx] = s_new[idx] + (qua + tri) * p1 + p2;
#pragma unroll
  for (int d = 0; d < 3; ++d) {
    size_t o = ((size_t)n * 3 + d) * HID + c;
    out_v[o] = v_new[o] + v1[d] * p3;
  }
}

// ---------------------------------------------------------------------------
extern "C" void kernel_launch(void* const* d_in, const int* in_sizes, int n_in,
                              void* d_out, int out_size, void* d_ws, size_t ws_size,
                              hipStream_t stream) {
  const float* node_scalar = (const float*)d_in[0];
  const float* node_vector = (const float*)d_in[1];
  const int*   edge_index  = (const int*)d_in[2];
  const float* dist        = (const float*)d_in[3];
  const float* edge_feats  = (const float*)d_in[4];
  const float* edge_vector = (const float*)d_in[5];
  const float* ln_gamma    = (const float*)d_in[6];
  const float* ln_beta     = (const float*)d_in[7];
  const float* alpha       = (const float*)d_in[8];
  const float* W_cross     = (const float*)d_in[9];
  const float* W_node      = (const float*)d_in[10];
  const float* b_node      = (const float*)d_in[11];
  const float* W_edge      = (const float*)d_in[12];
  const float* b_edge      = (const float*)d_in[13];
  const float* W_p1        = (const float*)d_in[14];
  const float* b_p1        = (const float*)d_in[15];
  const float* W_p2        = (const float*)d_in[16];
  const float* b_p2        = (const float*)d_in[17];
  const float* W_vec       = (const float*)d_in[18];
  const float* W_f         = (const float*)d_in[19];
  const float* b_f         = (const float*)d_in[20];
  (void)in_sizes; (void)n_in; (void)out_size; (void)ws_size;

  float* out_s = (float*)d_out;
  float* out_v = out_s + (size_t)N_NODES * HID;
  float* out_e = out_v + (size_t)N_NODES * 3 * HID;

  // Workspace layout (bytes).  pm (E x 256 f32) aliases ef|fgate since their
  // lifetimes end before pm is produced.  Total ~= 560 MB.
  char* wsb  = (char*)d_ws;
  size_t off = 0;
  auto alloc = [&](size_t bytes) {
    void* p = wsb + off;
    off += (bytes + 255) & ~(size_t)255;
    return p;
  };
  __bf16* lnbf  = (__bf16*)alloc((size_t)N_NODES * HID * 2);
  float*  ns    = (float*) alloc((size_t)N_NODES * HID * 4);
  float*  nv    = (float*) alloc((size_t)N_NODES * 3 * HID * 4);
  float*  snew  = (float*) alloc((size_t)N_NODES * HID * 4);
  float*  vnew  = (float*) alloc((size_t)N_NODES * 3 * HID * 4);
  __bf16* msgbf = (__bf16*)alloc((size_t)N_EDGES * HID * 2);
  float*  vv    = (float*) alloc((size_t)N_NODES * 3 * 256 * 4);
  float*  pp    = (float*) alloc((size_t)N_NODES * 384 * 4);
  float*  pm    = (float*) alloc((size_t)N_EDGES * 256 * 4);
  float*  ef    = pm;                           // alias (dead before pm written)
  float*  fgate = pm + (size_t)N_EDGES * HID;   // alias (dead before pm written)

  dim3 B(256);
  k_layernorm<<<(N_NODES + 7) / 8, B, 0, stream>>>(node_scalar, ln_gamma, ln_beta, lnbf);
  k_gemm_bA<128, true,  false><<<(N_NODES + 127) / 128, B, 0, stream>>>(lnbf, W_node, b_node, ns, N_NODES);
  k_gemm<128, true,  true ><<<(N_EDGES + 127) / 128, B, 0, stream>>>(edge_feats, W_edge, b_edge, ef, N_EDGES);
  k_gemm<128, false, false><<<(3 * N_NODES + 127) / 128, B, 0, stream>>>(node_vector, W_cross, nullptr, nv, 3 * N_NODES);
  k_gemm<128, true,  true ><<<(N_EDGES + 127) / 128, B, 0, stream>>>(edge_feats, W_f, b_f, fgate, N_EDGES);
  k_init<<<2048, B, 0, stream>>>(node_scalar, node_vector, snew, vnew);
  k_attn_msg<<<(N_EDGES + 7) / 8, B, 0, stream>>>(edge_index, dist, ns, ef, alpha, msgbf, snew);
  k_edge_update<<<(int)(((size_t)N_EDGES * HID + 255) / 256), B, 0, stream>>>(edge_index, edge_vector, nv, fgate, edge_feats, out_e);
  k_gemm_bA<256, true,  true ><<<(N_EDGES + 127) / 128, B, 0, stream>>>(msgbf, W_p1, b_p1, pm, N_EDGES);
  k_vmsg<<<(N_EDGES + 7) / 8, B, 0, stream>>>(edge_index, node_vector, edge_vector, pm, vnew);
  k_gemm<256, false, false><<<(3 * N_NODES + 127) / 128, B, 0, stream>>>(vnew, W_vec, nullptr, vv, 3 * N_NODES);
  k_gemm<384, true,  false><<<(N_NODES + 127) / 128, B, 0, stream>>>(snew, W_p2, b_p2, pp, N_NODES);
  k_node_final<<<(int)(((size_t)N_NODES * HID + 255) / 256), B, 0, stream>>>(snew, vnew, vv, pp, out_s, out_v);
}
